// MambaDecoder_4801773437303
// MI455X (gfx1250) — matrix operand order repaired and checked by reading, not compiled, not run on hardware
//
#include <hip/hip_runtime.h>
#include <cstdint>
#include <cstddef>

// ---------------------------------------------------------------------------
// Problem constants (from the reference)
// ---------------------------------------------------------------------------
constexpr int Bc = 2, Lc = 1024, DINc = 512, DMc = 1024, NLc = 4;
constexpr int Ec = 2048, Nst = 16, Rc = 64, Kcv = 4;
constexpr int Mc = Bc * Lc;          // 2048 fused (b,l) rows
constexpr int XD = Rc + 2 * Nst;     // 96 x_dbl columns

typedef __attribute__((ext_vector_type(16))) __bf16 bf16x16;
typedef __attribute__((ext_vector_type(8)))  float  f32x8;
typedef __attribute__((ext_vector_type(4)))  int    v4i;
typedef __attribute__((address_space(1)))    v4i    gv4i;   // global int4
typedef __attribute__((address_space(3)))    v4i    lv4i;   // LDS int4

#define DEVFN __device__ __forceinline__

// Use CDNA5 async global->LDS copies when the toolchain exposes them.
#if defined(__has_builtin)
#if __has_builtin(__builtin_amdgcn_global_load_async_to_lds_b128) && \
    __has_builtin(__builtin_amdgcn_s_wait_asynccnt)
#define USE_ASYNC_LDS 1
#endif
#endif
#ifndef USE_ASYNC_LDS
#define USE_ASYNC_LDS 0
#endif

DEVFN unsigned short f2bf(float f) {            // round-to-nearest-even f32->bf16
  uint32_t u = __float_as_uint(f);
  uint32_t r = (u + 0x7FFFu + ((u >> 16) & 1u)) >> 16;
  return (unsigned short)r;
}
DEVFN float sigm(float x) { return 1.0f / (1.0f + __expf(-x)); }

// ---------------------------------------------------------------------------
// bf16 WMMA GEMM:  C[m,n] = sum_k A[m,k] * W[n,k]   (+ optional bias[n])
// A: MxK bf16 row-major (lda), W: NxK bf16 row-major (ldb), C: MxN f32 (ldc)
//
// Block = 8 waves; block tile = 128 rows x (16*NSUB) cols. The W tile
// (16*NSUB rows x 32 k) is staged into double-buffered LDS once per k-step and
// shared by all 8 waves (8x traffic reduction); next tile is staged (async on
// CDNA5) while the current one feeds NSUB WMMAs per wave. Fragment layouts follow
// CDNA5 ISA 7.12.2 (16-bit A 16x32 / B 32x16, wave32).
// ---------------------------------------------------------------------------
template <int NSUB>
__global__ __launch_bounds__(256) void gemm_bf16_wmma(
    const unsigned short* __restrict__ A, int lda,
    const unsigned short* __restrict__ W, int ldb,
    float* __restrict__ C, int ldc,
    const float* __restrict__ bias, int M, int N, int K) {
  constexpr int NBLK = 16 * NSUB;          // block-wide column strip
  constexpr int LSTR = 40;                 // LDS row stride in elems (80B pad)
  constexpr int BUFE = NBLK * LSTR;        // elems per LDS buffer
  __shared__ unsigned short ldsB[2][BUFE];

  const int lane = threadIdx.x & 31;
  const int half = lane >> 4;              // K-half selector
  const int mr   = lane & 15;              // row (A) / col (B,C) within tile
  const int wave = threadIdx.x >> 5;       // 0..7
  const int tilesN = N / NBLK;
  const int mBase = (blockIdx.x / tilesN) * 128 + wave * 16;
  const int nBase = (blockIdx.x % tilesN) * NBLK;

  const unsigned short* aRow = A + (size_t)(mBase + mr) * lda;

  // Per-thread staging slot: one 16B chunk of the (NBLK x 32) W tile.
  const int srow = threadIdx.x >> 2;            // 0..63
  const int scol = (threadIdx.x & 3) * 8;       // element offset within k-step
  const bool doStage = (srow < NBLK);
  const unsigned short* sg = W + (size_t)(nBase + srow) * ldb + scol;
  unsigned short* sl = &ldsB[0][0] + srow * LSTR + scol;

  auto stage = [&](int buf, int k0) {
    if (doStage) {
#if USE_ASYNC_LDS
      __builtin_amdgcn_global_load_async_to_lds_b128(
          (gv4i*)(sg + k0), (lv4i*)(sl + buf * BUFE), 0, 0);
#else
      *(uint4*)(sl + buf * BUFE) = *(const uint4*)(sg + k0);
#endif
    }
  };
  auto publish = [&]() {
#if USE_ASYNC_LDS
    __builtin_amdgcn_s_wait_asynccnt(0);
#endif
    __syncthreads();
  };

  f32x8 acc[NSUB];
#pragma unroll
  for (int s = 0; s < NSUB; ++s)
#pragma unroll
    for (int i = 0; i < 8; ++i) acc[s][i] = 0.0f;

  union Frag { bf16x16 v; uint4 q[2]; };

  stage(0, 0);
  publish();

  int buf = 0;
  for (int k0 = 0; k0 < K; k0 += 32) {
    if (k0 + 32 < K) stage(buf ^ 1, k0 + 32);  // prefetch next tile into LDS

    Frag a;                                    // A 16x32 fragment (global)
    a.q[0] = *(const uint4*)(aRow + k0 + half * 8);
    a.q[1] = *(const uint4*)(aRow + k0 + 16 + half * 8);

#pragma unroll
    for (int s = 0; s < NSUB; ++s) {           // B 32x16 fragments from LDS
      const unsigned short* bp =
          &ldsB[buf][(s * 16 + mr) * LSTR + half * 8];
      Frag b;
      b.q[0] = *(const uint4*)bp;
      b.q[1] = *(const uint4*)(bp + 16);
      acc[s] = __builtin_amdgcn_wmma_f32_16x16x32_bf16(
          false, a.v, false, b.v, (short)0, acc[s], false, false);
    }
    publish();                                  // next tile staged, cur consumed
    buf ^= 1;
  }

#pragma unroll
  for (int s = 0; s < NSUB; ++s) {
    const int n = nBase + s * 16 + mr;
    const float bv = bias ? bias[n] : 0.0f;
#pragma unroll
    for (int i = 0; i < 8; ++i) {               // C: VGPR i -> M = i + 8*half
      const int m = mBase + half * 8 + i;
      C[(size_t)m * ldc + n] = acc[s][i] + bv;
    }
  }
}

// ---------------------------------------------------------------------------
// Elementwise / helper kernels
// ---------------------------------------------------------------------------
__global__ void cast_f32_bf16_kernel(const float* __restrict__ in,
                                     unsigned short* __restrict__ out, size_t n) {
  size_t i = (size_t)blockIdx.x * blockDim.x + threadIdx.x;
  if (i < n) out[i] = f2bf(in[i]);
}

// LayerNorm over D=1024 per row, bf16 output for the next WMMA GEMM.
__global__ __launch_bounds__(256) void layernorm_bf16_kernel(
    const float* __restrict__ x, const float* __restrict__ w,
    const float* __restrict__ b, unsigned short* __restrict__ out, int D) {
  __shared__ float red[16];
  const int row = blockIdx.x;
  const float* xr = x + (size_t)row * D;
  float s = 0.f, s2 = 0.f;
  for (int i = threadIdx.x; i < D; i += blockDim.x) {
    float v = xr[i]; s += v; s2 += v * v;
  }
  for (int off = 16; off > 0; off >>= 1) {
    s  += __shfl_down(s,  off);
    s2 += __shfl_down(s2, off);
  }
  const int wv = threadIdx.x >> 5, ln = threadIdx.x & 31;
  if (ln == 0) { red[wv] = s; red[8 + wv] = s2; }
  __syncthreads();
  if (threadIdx.x == 0) {
    float a = 0.f, a2 = 0.f;
    for (int i = 0; i < 8; ++i) { a += red[i]; a2 += red[8 + i]; }
    red[0] = a; red[1] = a2;
  }
  __syncthreads();
  const float mean = red[0] / D;
  const float var  = red[1] / D - mean * mean;
  const float rstd = rsqrtf(var + 1e-5f);
  for (int i = threadIdx.x; i < D; i += blockDim.x) {
    float v = (xr[i] - mean) * rstd * w[i] + b[i];
    out[(size_t)row * D + i] = f2bf(v);
  }
}

// Causal depthwise conv (K=4) + bias + SiLU over xm = xz[..., :E].
__global__ void conv_silu_kernel(const float* __restrict__ xz,
                                 const float* __restrict__ cw,
                                 const float* __restrict__ cb,
                                 float* __restrict__ uf,
                                 unsigned short* __restrict__ ub) {
  size_t idx = (size_t)blockIdx.x * blockDim.x + threadIdx.x;
  if (idx >= (size_t)Mc * Ec) return;
  const int e = (int)(idx % Ec);
  const size_t bl = idx / Ec;
  const int t = (int)(bl % Lc);
  const int b = (int)(bl / Lc);
  float acc = cb[e];
#pragma unroll
  for (int j = 0; j < Kcv; ++j) {
    const int tt = t - (Kcv - 1) + j;
    if (tt >= 0)
      acc += xz[((size_t)b * Lc + tt) * (2 * Ec) + e] * cw[e * Kcv + j];
  }
  const float sv = acc * sigm(acc);   // SiLU
  uf[idx] = sv;
  ub[idx] = f2bf(sv);
}

// Compact x_dbl[:, :R] into bf16 MxR for the dtproj GEMM.
__global__ void slice_cast_kernel(const float* __restrict__ xdbl,
                                  unsigned short* __restrict__ out) {
  size_t i = (size_t)blockIdx.x * blockDim.x + threadIdx.x;
  if (i >= (size_t)Mc * Rc) return;
  const size_t r = i / Rc, c = i % Rc;
  out[i] = f2bf(xdbl[r * XD + c]);
}

// softplus(g + dtproj_b[e]) in place.
__global__ void dt_softplus_kernel(float* __restrict__ g,
                                   const float* __restrict__ dtb) {
  size_t i = (size_t)blockIdx.x * blockDim.x + threadIdx.x;
  if (i >= (size_t)Mc * Ec) return;
  const int e = (int)(i % Ec);
  const float x = g[i] + dtb[e];
  g[i] = (x > 20.0f) ? x : log1pf(__expf(x));
}

// Selective scan: one lane per (b,e) channel, 16-state recurrence in VGPRs.
// b is derived from blockIdx so the per-step B/C row addresses are wave-uniform
// (scalar-load friendly). Fuses +u*D and SiLU(z) gating; bf16 output feeds the
// outproj WMMA GEMM.
__global__ __launch_bounds__(256) void scan_kernel(
    const float* __restrict__ dt, const float* __restrict__ u,
    const float* __restrict__ xdbl, const float* __restrict__ xz,
    const float* __restrict__ A_log, const float* __restrict__ Dp,
    unsigned short* __restrict__ ybf) {
  const int b = blockIdx.x >> 3;                       // 8 blocks per batch
  const int e = ((blockIdx.x & 7) << 8) + threadIdx.x; // 0..E-1
  float A[Nst], h[Nst];
#pragma unroll
  for (int n = 0; n < Nst; ++n) {
    A[n] = -__expf(A_log[(size_t)e * Nst + n]);
    h[n] = 0.0f;
  }
  const float Dv = Dp[e];
  for (int t = 0; t < Lc; ++t) {
    const size_t off = (size_t)b * Lc + t;
    const float dtv = dt[off * Ec + e];
    const float uv  = u[off * Ec + e];
    const float du  = dtv * uv;
    const float* Bm = xdbl + off * XD + Rc;        // uniform address
    const float* Cm = Bm + Nst;                    // uniform address
    float acc = 0.0f;
#pragma unroll
    for (int n = 0; n < Nst; ++n) {
      const float dA = __expf(dtv * A[n]);
      h[n] = dA * h[n] + du * Bm[n];
      acc += h[n] * Cm[n];
    }
    float y = acc + uv * Dv;
    const float zg = xz[off * (2 * Ec) + Ec + e];
    y *= zg * sigm(zg);                            // * SiLU(z)
    ybf[off * Ec + e] = f2bf(y);
  }
}

__global__ void add_inplace_kernel(float* __restrict__ h,
                                   const float* __restrict__ d, size_t n) {
  size_t i = (size_t)blockIdx.x * blockDim.x + threadIdx.x;
  if (i < n) h[i] += d[i];
}

// ---------------------------------------------------------------------------
// Host side
// ---------------------------------------------------------------------------
template <int NSUB>
static void run_gemm(const unsigned short* A, int lda, const unsigned short* W,
                     int ldb, float* C, int ldc, const float* bias,
                     int M, int N, int K, hipStream_t s) {
  const int blocks = (M / 128) * (N / (16 * NSUB));  // exact grid, no tail
  gemm_bf16_wmma<NSUB><<<blocks, 256, 0, s>>>(A, lda, W, ldb, C, ldc, bias,
                                              M, N, K);
}

extern "C" void kernel_launch(void* const* d_in, const int* in_sizes, int n_in,
                              void* d_out, int out_size, void* d_ws, size_t ws_size,
                              hipStream_t stream) {
  (void)in_sizes; (void)n_in; (void)out_size; (void)ws_size;
  const float* x      = (const float*)d_in[0];
  /* d_in[1] = z : unused by the reference */
  const float* ip_w   = (const float*)d_in[2];
  const float* ip_b   = (const float*)d_in[3];
  const float* ln_w   = (const float*)d_in[4];
  const float* ln_b   = (const float*)d_in[5];
  const float* inproj = (const float*)d_in[6];
  const float* conv_w = (const float*)d_in[7];
  const float* conv_b = (const float*)d_in[8];
  const float* xproj  = (const float*)d_in[9];
  const float* dtw    = (const float*)d_in[10];
  const float* dtb    = (const float*)d_in[11];
  const float* A_log  = (const float*)d_in[12];
  const float* Dp     = (const float*)d_in[13];
  const float* outw   = (const float*)d_in[14];
  const float* fn_w   = (const float*)d_in[15];
  const float* fn_b   = (const float*)d_in[16];
  const float* op_w   = (const float*)d_in[17];
  const float* op_b   = (const float*)d_in[18];
  float* out = (float*)d_out;

  // bump allocator over d_ws (deterministic; everything written before read)
  char* p = (char*)d_ws;
  auto alloc = [&](size_t bytes) -> void* {
    void* r = (void*)p; p += (bytes + 255) & ~(size_t)255; return r;
  };
  unsigned short* x_bf   = (unsigned short*)alloc((size_t)Mc * DINc * 2);
  unsigned short* wbuf   = (unsigned short*)alloc((size_t)2 * Ec * DMc * 2); // 8MB, reused
  float*          h      = (float*)alloc((size_t)Mc * DMc * 4);
  unsigned short* hn_bf  = (unsigned short*)alloc((size_t)Mc * DMc * 2);
  float*          xz     = (float*)alloc((size_t)Mc * 2 * Ec * 4);
  float*          u      = (float*)alloc((size_t)Mc * Ec * 4);
  unsigned short* u_bf   = (unsigned short*)alloc((size_t)Mc * Ec * 2);
  float*          xdbl   = (float*)alloc((size_t)Mc * XD * 4);
  unsigned short* dtA_bf = (unsigned short*)alloc((size_t)Mc * Rc * 2);
  float*          dtbuf  = (float*)alloc((size_t)Mc * Ec * 4);
  unsigned short* y_bf   = (unsigned short*)alloc((size_t)Mc * Ec * 2);
  float*          mout   = (float*)alloc((size_t)Mc * DMc * 4);
  unsigned short* hf_bf  = (unsigned short*)alloc((size_t)Mc * DMc * 2);

  auto cast = [&](const float* src, unsigned short* dst, size_t n) {
    cast_f32_bf16_kernel<<<(unsigned)((n + 255) / 256), 256, 0, stream>>>(src, dst, n);
  };

  // h = x @ ip_w^T + ip_b
  cast(x, x_bf, (size_t)Mc * DINc);
  cast(ip_w, wbuf, (size_t)DMc * DINc);
  run_gemm<4>(x_bf, DINc, wbuf, DINc, h, DMc, ip_b, Mc, DMc, DINc, stream);

  const unsigned EW = (unsigned)(((size_t)Mc * Ec + 255) / 256);
  for (int l = 0; l < NLc; ++l) {
    layernorm_bf16_kernel<<<Mc, 256, 0, stream>>>(h, ln_w + l * DMc,
                                                  ln_b + l * DMc, hn_bf, DMc);
    // xz = hn @ inproj^T  (2048 x 4096 x 1024)
    cast(inproj + (size_t)l * 2 * Ec * DMc, wbuf, (size_t)2 * Ec * DMc);
    run_gemm<4>(hn_bf, DMc, wbuf, DMc, xz, 2 * Ec, nullptr, Mc, 2 * Ec, DMc, stream);
    // u = silu(causal dwconv(xm) + cb)
    conv_silu_kernel<<<EW, 256, 0, stream>>>(xz, conv_w + (size_t)l * Ec * Kcv,
                                             conv_b + l * Ec, u, u_bf);
    // x_dbl = u @ xproj^T  (N = 96)
    cast(xproj + (size_t)l * XD * Ec, wbuf, (size_t)XD * Ec);
    run_gemm<3>(u_bf, Ec, wbuf, Ec, xdbl, XD, nullptr, Mc, XD, Ec, stream);
    // dt = softplus(x_dbl[:, :R] @ dtw^T + dtb)
    slice_cast_kernel<<<(unsigned)(((size_t)Mc * Rc + 255) / 256), 256, 0, stream>>>(xdbl, dtA_bf);
    cast(dtw + (size_t)l * Ec * Rc, wbuf, (size_t)Ec * Rc);
    run_gemm<4>(dtA_bf, Rc, wbuf, Rc, dtbuf, Ec, nullptr, Mc, Ec, Rc, stream);
    dt_softplus_kernel<<<EW, 256, 0, stream>>>(dtbuf, dtb + l * Ec);
    // selective scan (+ u*D, * silu(z)), bf16 output
    scan_kernel<<<(Bc * Ec) / 256, 256, 0, stream>>>(
        dtbuf, u, xdbl, xz, A_log + (size_t)l * Ec * Nst, Dp + l * Ec, y_bf);
    // mout = y @ outproj^T ; h += mout
    cast(outw + (size_t)l * DMc * Ec, wbuf, (size_t)DMc * Ec);
    run_gemm<4>(y_bf, Ec, wbuf, Ec, mout, DMc, nullptr, Mc, DMc, Ec, stream);
    add_inplace_kernel<<<(unsigned)(((size_t)Mc * DMc + 255) / 256), 256, 0, stream>>>(
        h, mout, (size_t)Mc * DMc);
  }

  // final LN + output projection (bias epilogue) straight into d_out
  layernorm_bf16_kernel<<<Mc, 256, 0, stream>>>(h, fn_w, fn_b, hf_bf, DMc);
  cast(op_w, wbuf, (size_t)DINc * DMc);
  run_gemm<4>(hf_bf, DMc, wbuf, DMc, out, DINc, op_b, Mc, DINc, DMc, stream);
}
